// Correlation_36730560316063
// MI455X (gfx1250) — compile-verified, matrix operand order checked by reading, not run on hardware
//
#include <hip/hip_runtime.h>

// ---------------------------------------------------------------------------
// FlowNet-style correlation on MI455X (gfx1250, wave32, WMMA f16).
//
// out[b, i*41+j, y, p] = (1/576) * sum_c in1[b,c,y,p] * in2[b,c,y-dy_j,p-dx_i]
//   dx_i = 2i-20, dy_j = 2j-20, i,j in [0,41)
//
// Per (b, y, dy): full row-pair correlation full[p,q] = sum_c in1[p,c]in2[q,c]
// as f16 WMMA GEMMs (K=64 = 2x32, f32 accumulate), scattering the 41 even
// diagonals (i = (p-q+20)/2). dx is always even -> parity-split pixels so
// every computed (p,q) has matching parity.
//
// Block = (b, y, j-slice): 320 threads = 10 waves = 5 M-tiles x 2 parities.
// A (16 px x 64 ch, pre-scaled by 1/576) lives in VGPRs for the whole kernel.
// Per dy, the in2 row is staged into LDS pre-swizzled into the WMMA B operand
// layout: float4 global loads + v_cvt_pk_rtz_f16_f32 packing channel pairs
// into single ds_store_b32. Compute waves read operands as 2x ds_load_b128.
// Output stores are non-temporal (413 MB output >> 192 MB L2), addressed as
// saddr + 32-bit scaled index + immediate offsets; the 8-store scatter uses
// two pinned 32-bit base indices so every immediate fits signed 24 bits.
// ---------------------------------------------------------------------------

typedef __attribute__((ext_vector_type(16))) _Float16 v16h;
typedef __attribute__((ext_vector_type(8)))  float    v8f;

#define B_   4
#define C_   64
#define H_   96
#define W_   160
#define DD   41
#define HW   (H_*W_)
#define CHW  (C_*HW)
#define SCALE (1.0f/576.0f)
#define JSPLIT 4
#define CSTRIDE (DD*HW)        // offset delta when i += 1
#define RSTEP   (DD*HW + 2)    // offset delta when r += 1 (i+=1, p+=2)
#define TSTEP   (16*DD*HW)     // offset delta when t += 1 (i-=16)

__global__ __launch_bounds__(320)
void corr_wmma_kernel(const float* __restrict__ in1,
                      const float* __restrict__ in2,
                      float* __restrict__ out) {
    const int y  = blockIdx.x;   // 0..95
    const int b  = blockIdx.y;   // 0..3
    const int jz = blockIdx.z;   // 0..JSPLIT-1

    const int tid  = threadIdx.x;
    const int lane = tid & 31;
    const int wv   = tid >> 5;   // wave 0..9
    const int pa   = wv & 1;     // pixel parity handled by this wave
    const int wi   = wv >> 1;    // tile index within parity, 0..4
    const int g    = lane >> 4;
    const int n_l  = lane & 15;

    // B staging, pre-swizzled to WMMA B operand layout (u32 = packed f16 pair):
    // [parity(2)][qtile(5)][kchunk(2)][lane(32)][8 dwords] = 20 KB
    __shared__ __attribute__((aligned(32))) unsigned BsU[2*5*2*32*8];
    const v16h* Bv = (const v16h*)BsU;

    // ---- A tile into registers (held for whole kernel), pre-scaled ----
    // ISA 16-bit A layout: lane l: m=l&15, g=l>>4;
    //   element (v,h): kk = (v>=4?16:0) + g*8 + 2*(v&3) + h
    const int p_a = pa + 2*(16*wi + n_l);
    const float* a_base = in1 + (size_t)b*CHW + (size_t)y*W_ + p_a;
    union { unsigned u[8]; v16h v; } A0, A1;
    #pragma unroll
    for (int v = 0; v < 8; ++v) {
        int kk = ((v >= 4) ? 16 : 0) + g*8 + 2*(v & 3);   // h=0 channel
        float f00 = a_base[(size_t)kk      * HW] * SCALE;
        float f01 = a_base[(size_t)(kk+1)  * HW] * SCALE;
        float f10 = a_base[(size_t)(kk+32) * HW] * SCALE;
        float f11 = a_base[(size_t)(kk+33) * HW] * SCALE;
        A0.u[v] = __builtin_bit_cast(unsigned, __builtin_amdgcn_cvt_pkrtz(f00, f01));
        A1.u[v] = __builtin_bit_cast(unsigned, __builtin_amdgcn_cvt_pkrtz(f10, f11));
    }

    // ---- staging decomposition (invariant over j) ----
    const int m_q = tid % 40;            // q quad index: q = 4*m_q + e
    const int cp_t = tid / 40;           // 0..7 channel-pair slot
    const int t_e0 = (2*m_q)     >> 4, n_e0 = (2*m_q)     & 15;  // e = 0,1
    const int t_e1 = (2*m_q + 1) >> 4, n_e1 = (2*m_q + 1) & 15;  // e = 2,3

    // ---- output offset bases (affine in j, t, r) ----
    const int i0base = 16*wi + g*8 + 10 - n_l;      // i = i0base - 16t + r
    const int p0     = pa + 2*(16*wi + g*8);        // p = p0 + 2r
    int offj = b*(DD*DD*HW) + y*W_ + i0base*CSTRIDE + p0 + jz*HW;

    for (int j = jz; j < DD; j += JSPLIT) {
        const int yy = y - 2*j + 20;                // y' = y - dy_j
        const bool rowok = (yy >= 0) && (yy < H_);

        // ---- stage B = in2[b, :, y', :] into swizzled LDS ----
        if (rowok) {
            const float* src = in2 + (size_t)b*CHW + (size_t)yy*W_ + 4*m_q;
            #pragma unroll
            for (int it = 0; it < 4; ++it) {
                int c  = 2*(it*8 + cp_t);           // even channel
                float4 r0 = *(const float4*)(src + (size_t)c*HW);
                float4 r1 = *(const float4*)(src + (size_t)(c+1)*HW);
                int kc = c >> 5, kk = c & 31;
                int lg = ((kk >> 3) & 1) * 16;
                int vv = ((kk >> 4) << 2) | ((kk >> 1) & 3);
                unsigned pk0 = __builtin_bit_cast(unsigned, __builtin_amdgcn_cvt_pkrtz(r0.x, r1.x));
                unsigned pk1 = __builtin_bit_cast(unsigned, __builtin_amdgcn_cvt_pkrtz(r0.y, r1.y));
                unsigned pk2 = __builtin_bit_cast(unsigned, __builtin_amdgcn_cvt_pkrtz(r0.z, r1.z));
                unsigned pk3 = __builtin_bit_cast(unsigned, __builtin_amdgcn_cvt_pkrtz(r0.w, r1.w));
                BsU[((((0*5 + t_e0)*2 + kc)*32 + lg + n_e0)*8) + vv] = pk0;
                BsU[((((1*5 + t_e0)*2 + kc)*32 + lg + n_e0)*8) + vv] = pk1;
                BsU[((((0*5 + t_e1)*2 + kc)*32 + lg + n_e1)*8) + vv] = pk2;
                BsU[((((1*5 + t_e1)*2 + kc)*32 + lg + n_e1)*8) + vv] = pk3;
            }
        }
        // prefetch next dy's in2 row into cache while we compute this one
        {
            int jn = j + JSPLIT;
            int yyn = y - 2*jn + 20;
            if (jn < DD && yyn >= 0 && yyn < H_) {
                const float* pf = in2 + (size_t)b*CHW + (size_t)yyn*W_
                                  + (size_t)(tid/5)*HW + (tid%5)*32;
                __builtin_prefetch(pf, 0, 3);
            }
        }
        __syncthreads();

        // ---- compute 4-tile sliding q-window + scatter diagonals ----
        #pragma unroll
        for (int tr = 0; tr < 4; ++tr) {
            const int t = wi - 2 + tr;              // virtual q-tile index
            v8f d = {0.f,0.f,0.f,0.f,0.f,0.f,0.f,0.f};
            if (rowok && t >= 0 && t <= 4) {
                const int bb = ((pa*5 + t)*2)*32 + lane;
                v16h b0 = Bv[bb];
                v16h b1 = Bv[bb + 32];
                d = __builtin_amdgcn_wmma_f32_16x16x32_f16(
                        false, A0.v, false, b0, (short)0, d, false, false);
                d = __builtin_amdgcn_wmma_f32_16x16x32_f16(
                        false, A1.v, false, b1, (short)0, d, false, false);
            }
            const int i_t    = i0base - 16*t;
            const int off_lo = offj - t*TSTEP;      // base index for r = 0..3
            int off_hi = off_lo + 4*RSTEP;          // base index for r = 4..7
            // Pin off_hi in a VGPR and hide its derivation from address
            // folding, so r=4..7 stores select the saddr + 32-bit scaled
            // index form with in-range immediates instead of rebuilding a
            // 64-bit address per store.
            asm("" : "+v"(off_hi));
            #pragma unroll
            for (int r = 0; r < 4; ++r) {           // immediates <= 3*RSTEP*4
                if ((unsigned)(i_t + r) <= 40u) {
                    __builtin_nontemporal_store(d[r], out + (off_lo + r*RSTEP));
                }
            }
            #pragma unroll
            for (int r = 0; r < 4; ++r) {           // immediates <= 3*RSTEP*4
                if ((unsigned)(i_t + 4 + r) <= 40u) {
                    __builtin_nontemporal_store(d[4 + r], out + (off_hi + r*RSTEP));
                }
            }
        }
        offj += JSPLIT*HW;
        __syncthreads();
    }
}

extern "C" void kernel_launch(void* const* d_in, const int* in_sizes, int n_in,
                              void* d_out, int out_size, void* d_ws, size_t ws_size,
                              hipStream_t stream) {
    (void)in_sizes; (void)n_in; (void)out_size; (void)d_ws; (void)ws_size;
    const float* in1 = (const float*)d_in[0];
    const float* in2 = (const float*)d_in[1];
    float* out = (float*)d_out;
    dim3 grid(H_, B_, JSPLIT);   // (y, b, j-slice)
    corr_wmma_kernel<<<grid, 320, 0, stream>>>(in1, in2, out);
}
